// TransformerEncoderLayer_22574348108467
// MI455X (gfx1250) — matrix operand order repaired and checked by reading, not compiled, not run on hardware
//
#include <hip/hip_runtime.h>
#include <hip/hip_bf16.h>

// ---------------- types ----------------
typedef __bf16 bf16;
typedef __attribute__((ext_vector_type(16))) __bf16 bf16x16;
typedef __attribute__((ext_vector_type(8)))  __bf16 bf16x8;
typedef __attribute__((ext_vector_type(8)))  float  f32x8;
typedef __attribute__((ext_vector_type(4)))  unsigned u32x4;
typedef __attribute__((ext_vector_type(4)))  int      i32x4;
typedef __attribute__((ext_vector_type(8)))  int      i32x8;

union VecU { bf16x16 f; bf16x8 h[2]; };

#define B_  8
#define N_  1024
#define C_  1024
#define H_  16
#define D_  64
#define HID_ 4096
#define BN_ (B_ * N_)          // 8192 rows
#define EPS_ 1e-5f

#if defined(__has_builtin)
#  if __has_builtin(__builtin_amdgcn_tensor_load_to_lds) && __has_builtin(__builtin_amdgcn_s_wait_tensorcnt)
#    define HAVE_TDM 1
#  endif
#endif
#ifndef HAVE_TDM
#  define HAVE_TDM 0
#endif

// LDS row strides (elements) after TDM padding
#define A_STRIDE  40   // 32 elems (64B) + 16B pad  -> 80B rows
#define W_STRIDE  136  // 128 elems (256B) + 16B pad -> 272B rows

// pad cfg bits for D# group1 word0: pad_enable | pad_interval | pad_amount
// interval code k -> 2^(k+1) DWORDs before pad; amount code a -> (a+1) DWORDs pad
#define PADCFG_A  ((1u << 20) | (3u << 22) | (3u << 25))  // every 16 DW (64B), pad 4 DW (16B)
#define PADCFG_W  ((1u << 20) | (5u << 22) | (3u << 25))  // every 64 DW (256B), pad 4 DW (16B)

#if HAVE_TDM
// Issue a 2D tensor-tile DMA: global (row-major, row_stride elems) -> LDS.
__device__ __forceinline__ void tdm_load_2d(unsigned lds_off, const bf16* gptr,
                                            unsigned tile_d0, unsigned tile_d1,
                                            unsigned tensor_d0, unsigned tensor_d1,
                                            unsigned row_stride, unsigned pad_cfg) {
    unsigned long long ga = (unsigned long long)(uintptr_t)gptr;
    u32x4 g0;
    g0[0] = 1u;                                     // count=1 valid descriptor
    g0[1] = lds_off;                                // lds_addr (bytes)
    g0[2] = (unsigned)ga;                           // global_addr[31:0]
    g0[3] = (unsigned)(ga >> 32) | (2u << 30);      // global_addr[56:32] | type=2
    i32x8 g1;
    g1[0] = (int)((1u << 16) | pad_cfg);            // data_size=1 (2B) | padding
    g1[1] = (int)((tensor_d0 & 0xFFFFu) << 16);     // tensor_dim0[15:0]
    g1[2] = (int)(((tensor_d0 >> 16) & 0xFFFFu) | ((tensor_d1 & 0xFFFFu) << 16));
    g1[3] = (int)(((tensor_d1 >> 16) & 0xFFFFu) | (tile_d0 << 16));  // tile_dim0
    g1[4] = (int)(tile_d1 & 0xFFFFu);               // tile_dim1 (tile_dim2=0)
    g1[5] = (int)row_stride;                        // tensor_dim0_stride[31:0]
    g1[6] = 0;                                      // stride hi / dim1_stride lo
    g1[7] = 0;
    i32x4 z4 = {0, 0, 0, 0};
    i32x8 z8 = {0, 0, 0, 0, 0, 0, 0, 0};
    __builtin_amdgcn_tensor_load_to_lds(g0, g1, z4, z4, z8, 0);
}
#endif

// ---------------- fp32 [N,K] -> bf16 transposed [K,N] ----------------
__global__ void f32_to_bf16_T_kernel(const float* __restrict__ src,
                                     bf16* __restrict__ dst, int Nrows, int Kcols) {
    int i = blockIdx.x * 256 + threadIdx.x;
    if (i < Nrows * Kcols) {
        int n = i / Kcols, k = i % Kcols;
        dst[(size_t)k * Nrows + n] = (bf16)src[i];
    }
}

// ---------------- LayerNorm (row of 1024) -> bf16 ----------------
__global__ __launch_bounds__(256) void layernorm_kernel(
    const float* __restrict__ x, const float* __restrict__ w,
    const float* __restrict__ b, bf16* __restrict__ y) {
    __shared__ float red[8];
    const int row = blockIdx.x;
    const int tid = threadIdx.x;
    const float* xr = x + (size_t)row * C_;
    float v[4];
    float s = 0.f;
#pragma unroll
    for (int i = 0; i < 4; i++) { v[i] = xr[tid + i * 256]; s += v[i]; }
#pragma unroll
    for (int d = 1; d < 32; d <<= 1) s += __shfl_xor(s, d, 32);
    if ((tid & 31) == 0) red[tid >> 5] = s;
    __syncthreads();
    float mean = 0.f;
#pragma unroll
    for (int i = 0; i < 8; i++) mean += red[i];
    mean *= (1.0f / (float)C_);
    __syncthreads();
    s = 0.f;
#pragma unroll
    for (int i = 0; i < 4; i++) { float d0 = v[i] - mean; s += d0 * d0; }
#pragma unroll
    for (int d = 1; d < 32; d <<= 1) s += __shfl_xor(s, d, 32);
    if ((tid & 31) == 0) red[tid >> 5] = s;
    __syncthreads();
    float var = 0.f;
#pragma unroll
    for (int i = 0; i < 8; i++) var += red[i];
    var *= (1.0f / (float)C_);
    const float rstd = rsqrtf(var + EPS_);
    bf16* yr = y + (size_t)row * C_;
#pragma unroll
    for (int i = 0; i < 4; i++) {
        int c = tid + i * 256;
        yr[c] = (bf16)((v[i] - mean) * rstd * w[c] + b[c]);
    }
}

// ---------------- bf16 WMMA GEMM: Y = X @ W^T (+bias)(+gelu) ------------------
// X:  [M,K] bf16 row-major.  WT: [K,N] bf16 row-major (pre-transposed weights).
// Block tile 256x128, 8 waves (4Mx2N), wave tile 64x64, K-step 32.
// Tiles staged to LDS by the Tensor Data Mover (double-buffered, padded rows).
template<int OUT_BF16, int DO_GELU, int DO_BIAS>
__global__ __launch_bounds__(256) void gemm_kernel(
    const bf16* __restrict__ X, const bf16* __restrict__ WT,
    const float* __restrict__ bias, void* __restrict__ Yv,
    int M, int N, int K) {
    __shared__ bf16 ldsA[2][256 * A_STRIDE];
    __shared__ bf16 ldsW[2][32 * W_STRIDE];
    const int tid  = threadIdx.x;
    const int wid  = tid >> 5;
    const int lane = tid & 31;
    const int waveM = wid & 3;              // 4 waves in M (64 rows each)
    const int waveN = wid >> 2;             // 2 waves in N (64 cols each)
    const int m0 = blockIdx.y * 256;
    const int n0 = blockIdx.x * 128;
    const int half = lane >> 4;
    const int lr   = lane & 15;

    f32x8 c[4][4];
#pragma unroll
    for (int r = 0; r < 4; r++)
#pragma unroll
        for (int ci = 0; ci < 4; ci++)
#pragma unroll
            for (int i = 0; i < 8; i++) c[r][ci][i] = 0.f;

#if HAVE_TDM
    unsigned ldsA_off[2] = { (unsigned)(uintptr_t)&ldsA[0][0],
                             (unsigned)(uintptr_t)&ldsA[1][0] };
    unsigned ldsW_off[2] = { (unsigned)(uintptr_t)&ldsW[0][0],
                             (unsigned)(uintptr_t)&ldsW[1][0] };
#endif

    auto stage = [&](int buf, int k0) {
#if HAVE_TDM
        if (wid == 0) {
            // A tile: 256 rows x 32 k (row-major in X)
            tdm_load_2d(ldsA_off[buf], X + (size_t)m0 * K + k0,
                        32u, 256u, (unsigned)K, (unsigned)M, (unsigned)K, PADCFG_A);
            // W^T tile: 32 k-rows x 128 n (row-major in WT)
            tdm_load_2d(ldsW_off[buf], WT + (size_t)k0 * N + n0,
                        128u, 32u, (unsigned)N, (unsigned)K, (unsigned)N, PADCFG_W);
        }
#else
        {   // cooperative fallback: same padded LDS layout
#pragma unroll
            for (int rep = 0; rep < 2; rep++) {
                int row = rep * 128 + (tid >> 1), hf = tid & 1;
                bf16x16 xa = *(const bf16x16*)(X + (size_t)(m0 + row) * K + k0 + hf * 16);
                *(bf16x16*)&ldsA[buf][row * A_STRIDE + hf * 16] = xa;
            }
            int wr = tid >> 3, seg = tid & 7;
            bf16x16 wv = *(const bf16x16*)(WT + (size_t)(k0 + wr) * N + n0 + seg * 16);
            *(bf16x16*)&ldsW[buf][wr * W_STRIDE + seg * 16] = wv;
        }
#endif
    };

    stage(0, 0);
    int it = 0;
    for (int k0 = 0; k0 < K; k0 += 32, ++it) {
        const int buf = it & 1;
        __syncthreads();                      // all readers of buf^1 are done
        const bool more = (k0 + 32) < K;
        if (more) stage(buf ^ 1, k0 + 32);    // DMA next tile while computing
#if HAVE_TDM
        if (wid == 0) {
            if (more) __builtin_amdgcn_s_wait_tensorcnt(2);  // current tile landed
            else      __builtin_amdgcn_s_wait_tensorcnt(0);
        }
#endif
        __syncthreads();                      // release: current tile ready

        // A fragments (16x32 bf16: lane=row, elems span K per ISA layout)
        bf16x16 a[4];
#pragma unroll
        for (int r = 0; r < 4; r++) {
            int m = waveM * 64 + r * 16 + lr;
            VecU u;
            u.h[0] = *(const bf16x8*)&ldsA[buf][m * A_STRIDE + half * 8];
            u.h[1] = *(const bf16x8*)&ldsA[buf][m * A_STRIDE + 16 + half * 8];
            a[r] = u.f;
        }
#pragma unroll
        for (int ci = 0; ci < 4; ci++) {
            // B fragment (32x16: lane = n col, elems span K per ISA layout) - gathered from W^T tile
            bf16x16 bfr;
#pragma unroll
            for (int s = 0; s < 16; s++)
                bfr[s] = ldsW[buf][((s & 7) + 8 * half + 16 * (s >> 3)) * W_STRIDE + waveN * 64 + ci * 16 + lr];
#pragma unroll
            for (int r = 0; r < 4; r++)
                c[r][ci] = __builtin_amdgcn_wmma_f32_16x16x32_bf16(
                    false, a[r], false, bfr, (short)0, c[r][ci], false, false);
        }
    }

#pragma unroll
    for (int r = 0; r < 4; r++)
#pragma unroll
        for (int ci = 0; ci < 4; ci++)
#pragma unroll
            for (int vv = 0; vv < 8; vv++) {
                int m = m0 + waveM * 64 + r * 16 + vv + 8 * half;
                int n = n0 + waveN * 64 + ci * 16 + lr;
                float val = c[r][ci][vv];
                if (DO_BIAS) val += bias[n];
                if (DO_GELU) val = 0.5f * val * (1.0f + erff(val * 0.70710678f));
                if (OUT_BF16) ((bf16*)Yv)[(size_t)m * N + n] = (bf16)val;
                else          ((float*)Yv)[(size_t)m * N + n] = val;
            }
}

// ---------------- split qkv -> Q*scale[BH,N,D], K^T[BH,D,N], V[BH,N,D] --------
__global__ void split_qkv_kernel(const bf16* __restrict__ qkv,
                                 bf16* __restrict__ q, bf16* __restrict__ kt,
                                 bf16* __restrict__ v) {
    int idx = blockIdx.x * 256 + threadIdx.x;      // < 8192*3072
    int col = idx % (3 * C_);
    int bn  = idx / (3 * C_);
    int s = col >> 10;
    int h = (col >> 6) & 15;
    int d = col & 63;
    int b = bn >> 10;
    int n = bn & 1023;
    int bh = b * H_ + h;
    bf16 val = qkv[idx];
    if (s == 0)      q [((size_t)bh * N_ + n) * D_ + d] = (bf16)((float)val * 0.125f);
    else if (s == 1) kt[((size_t)bh * D_ + d) * N_ + n] = val;
    else             v [((size_t)bh * N_ + n) * D_ + d] = val;
}

// ---------------- flash attention: one wave per 16 query rows -----------------
// 64 keys per step. Softmax denominator rides the matrix pipe: one extra WMMA
// against a constant ones-column B fragment accumulates row sums in o[4] col 0.
__global__ __launch_bounds__(256) void attention_kernel(
    const bf16* __restrict__ q, const bf16* __restrict__ kt,
    const bf16* __restrict__ v, bf16* __restrict__ out) {
    __shared__ bf16 ldsP[8][16 * 64];
    const int wid  = threadIdx.x >> 5;
    const int lane = threadIdx.x & 31;
    const int gw = blockIdx.x * 8 + wid;
    const int bh = gw >> 6;                 // / (N/16)
    const int qt = gw & 63;                 // query tile
    const bf16* qb  = q  + (size_t)bh * N_ * D_;
    const bf16* ktb = kt + (size_t)bh * D_ * N_;
    const bf16* vb  = v  + (size_t)bh * N_ * D_;
    const int half = lane >> 4;
    const int lr   = lane & 15;

    // Q A-fragments (16 rows x 64 d = two K=32 frags); Q is pre-scaled by D^-0.5
    bf16x16 aq[2];
    {
        int m = qt * 16 + lr;
#pragma unroll
        for (int kk = 0; kk < 2; kk++) {
            VecU u;
            u.h[0] = *(const bf16x8*)(qb + (size_t)m * D_ + kk * 32 + half * 8);
            u.h[1] = *(const bf16x8*)(qb + (size_t)m * D_ + kk * 32 + 16 + half * 8);
            aq[kk] = u.f;
        }
    }

    // constant ones-column B fragment: B[k][0]=1, rest 0
    bf16x16 bones;
#pragma unroll
    for (int i = 0; i < 16; i++) bones[i] = (bf16)((lr == 0) ? 1.0f : 0.0f);

    f32x8 o[5];                             // o[0..3]=P@V, o[4] col0 = row sums
    float mrow[8];
#pragma unroll
    for (int g = 0; g < 5; g++)
#pragma unroll
        for (int i = 0; i < 8; i++) o[g][i] = 0.f;
#pragma unroll
    for (int i = 0; i < 8; i++) mrow[i] = -1e30f;

    for (int j0 = 0; j0 < N_; j0 += 64) {
        if (j0 + 64 < N_) {                 // prefetch next K/V tiles
            __builtin_prefetch(ktb + (size_t)lane * N_ + j0 + 64, 0, 1);
            __builtin_prefetch(vb + (size_t)(j0 + 64 + lane) * D_, 0, 1);
            __builtin_prefetch(vb + (size_t)(j0 + 96 + lane) * D_, 0, 1);
        }
        // S = Q @ K^T for 64 keys (four 16-col C frags)
        f32x8 s[4];
#pragma unroll
        for (int hh = 0; hh < 4; hh++) {
            f32x8 acc;
#pragma unroll
            for (int i = 0; i < 8; i++) acc[i] = 0.f;
#pragma unroll
            for (int kk = 0; kk < 2; kk++) {
                bf16x16 bk;                 // B frag: lane = n col, elems span k
#pragma unroll
                for (int s = 0; s < 16; s++)
                    bk[s] = ktb[(size_t)(kk * 32 + (s & 7) + 8 * half + 16 * (s >> 3)) * N_ + j0 + hh * 16 + lr];
                acc = __builtin_amdgcn_wmma_f32_16x16x32_bf16(
                    false, aq[kk], false, bk, (short)0, acc, false, false);
            }
            s[hh] = acc;
        }
        // online row-max over 64 new columns; stage-major butterflies so the
        // 8 independent permutes of each stage can be clause-batched
        float cm[8];
#pragma unroll
        for (int vv = 0; vv < 8; vv++)
            cm[vv] = fmaxf(fmaxf(s[0][vv], s[1][vv]), fmaxf(s[2][vv], s[3][vv]));
#pragma unroll
        for (int st = 1; st <= 8; st <<= 1) {
            float t[8];
#pragma unroll
            for (int vv = 0; vv < 8; vv++) t[vv] = __shfl_xor(cm[vv], st, 32);
#pragma unroll
            for (int vv = 0; vv < 8; vv++) cm[vv] = fmaxf(cm[vv], t[vv]);
        }
        float fac[8];
#pragma unroll
        for (int vv = 0; vv < 8; vv++) {
            float mnew = fmaxf(mrow[vv], cm[vv]);
            fac[vv]  = __expf(mrow[vv] - mnew);
            mrow[vv] = mnew;
            int ml = vv + 8 * half;         // stash P (C layout) for A re-layout
#pragma unroll
            for (int hh = 0; hh < 4; hh++)
                ldsP[wid][ml * 64 + hh * 16 + lr] = (bf16)__expf(s[hh][vv] - mnew);
        }
        // reload P as two 16x32 A fragments (same-wave LDS, in-order)
        bf16x16 ap[2];
#pragma unroll
        for (int kk = 0; kk < 2; kk++) {
            VecU u;
            u.h[0] = *(const bf16x8*)&ldsP[wid][lr * 64 + kk * 32 + half * 8];
            u.h[1] = *(const bf16x8*)&ldsP[wid][lr * 64 + kk * 32 + 16 + half * 8];
            ap[kk] = u.f;
        }
        // O = O*fac + P @ V ; row-sum column rides along as a 5th fragment
#pragma unroll
        for (int g = 0; g < 5; g++) {
            f32x8 cc;
#pragma unroll
            for (int vv = 0; vv < 8; vv++) cc[vv] = o[g][vv] * fac[vv];
#pragma unroll
            for (int kk = 0; kk < 2; kk++) {
                bf16x16 bv;
                if (g < 4) {
#pragma unroll
                    for (int s = 0; s < 16; s++)
                        bv[s] = vb[(size_t)(j0 + kk * 32 + (s & 7) + 8 * half + 16 * (s >> 3)) * D_ + g * 16 + lr];
                }
                else       bv = bones;
                cc = __builtin_amdgcn_wmma_f32_16x16x32_bf16(
                    false, ap[kk], false, bv, (short)0, cc, false, false);
            }
            o[g] = cc;
        }
    }

    // broadcast per-row softmax denominator from column 0 (lanes 0 / 16)
    float lsum[8];
#pragma unroll
    for (int vv = 0; vv < 8; vv++) lsum[vv] = __shfl(o[4][vv], lane & 16, 32);

    const int b = bh >> 4, h = bh & 15;
#pragma unroll
    for (int g = 0; g < 4; g++)
#pragma unroll
        for (int vv = 0; vv < 8; vv++) {
            int m   = qt * 16 + vv + 8 * half;
            int col = h * D_ + g * 16 + lr;
            out[((size_t)(b * N_ + m)) * C_ + col] = (bf16)(o[g][vv] / lsum[vv]);
        }
}

// ---------------- residual: out = x + gamma * delta ---------------------------
__global__ void residual_kernel(const float* __restrict__ x,
                                const float* __restrict__ delta,
                                const float* __restrict__ gamma,
                                float* __restrict__ out, int total) {
    int i = blockIdx.x * 256 + threadIdx.x;
    if (i < total) out[i] = x[i] + gamma[i & (C_ - 1)] * delta[i];
}

// ---------------- launcher ----------------
extern "C" void kernel_launch(void* const* d_in, const int* in_sizes, int n_in,
                              void* d_out, int out_size, void* d_ws, size_t ws_size,
                              hipStream_t stream) {
    (void)in_sizes; (void)n_in; (void)out_size; (void)ws_size;
    const float* x      = (const float*)d_in[0];
    const float* ln1_w  = (const float*)d_in[1];
    const float* ln1_b  = (const float*)d_in[2];
    const float* qkv_w  = (const float*)d_in[3];
    const float* proj_w = (const float*)d_in[4];
    const float* proj_b = (const float*)d_in[5];
    const float* ln2_w  = (const float*)d_in[6];
    const float* ln2_b  = (const float*)d_in[7];
    const float* fc1_w  = (const float*)d_in[8];
    const float* fc1_b  = (const float*)d_in[9];
    const float* fc2_w  = (const float*)d_in[10];
    const float* fc2_b  = (const float*)d_in[11];
    const float* gamma1 = (const float*)d_in[12];
    const float* gamma2 = (const float*)d_in[13];
    float* out = (float*)d_out;

    char* ws = (char*)d_ws;
    size_t off = 0;
    auto alloc = [&](size_t bytes) -> void* {
        void* p = ws + off;
        off += (bytes + 255) & ~(size_t)255;
        return p;
    };
    // transposed bf16 weights: WT[k][n]
    bf16* wqkvT  = (bf16*)alloc((size_t)3 * C_ * C_ * 2);      // [C, 3C]   6 MB
    bf16* wprojT = (bf16*)alloc((size_t)C_ * C_ * 2);          // [C, C]    2 MB
    bf16* wfc1T  = (bf16*)alloc((size_t)HID_ * C_ * 2);        // [C, HID]  8 MB
    bf16* wfc2T  = (bf16*)alloc((size_t)C_ * HID_ * 2);        // [HID, C]  8 MB
    bf16* lnbuf  = (bf16*)alloc((size_t)BN_ * C_ * 2);         // 16 MB
    bf16* qkvbuf = (bf16*)alloc((size_t)BN_ * 3 * C_ * 2);     // 48 MB
    bf16* qb     = (bf16*)alloc((size_t)BN_ * C_ * 2);         // 16 MB
    bf16* ktb    = (bf16*)alloc((size_t)BN_ * C_ * 2);         // 16 MB
    bf16* vb     = (bf16*)alloc((size_t)BN_ * C_ * 2);         // 16 MB
    bf16* attno  = (bf16*)alloc((size_t)BN_ * C_ * 2);         // 16 MB
    float* projf = (float*)alloc((size_t)BN_ * C_ * 4);        // 32 MB
    float* x1    = (float*)alloc((size_t)BN_ * C_ * 4);        // 32 MB
    bf16*  hbuf  = qkvbuf;          // alias: qkvbuf+qb dead by fc1 (64 MB)
    float* fc2f  = projf;           // alias: projf dead after residual1

    // 1) weights fp32 -> bf16, transposed to [K,N]
    f32_to_bf16_T_kernel<<<(3 * C_ * C_ + 255) / 256, 256, 0, stream>>>(qkv_w, wqkvT, 3 * C_, C_);
    f32_to_bf16_T_kernel<<<(C_ * C_ + 255) / 256, 256, 0, stream>>>(proj_w, wprojT, C_, C_);
    f32_to_bf16_T_kernel<<<(HID_ * C_ + 255) / 256, 256, 0, stream>>>(fc1_w, wfc1T, HID_, C_);
    f32_to_bf16_T_kernel<<<(C_ * HID_ + 255) / 256, 256, 0, stream>>>(fc2_w, wfc2T, C_, HID_);

    // 2) LN1
    layernorm_kernel<<<BN_, 256, 0, stream>>>(x, ln1_w, ln1_b, lnbuf);

    // 3) qkv = ln1 @ qkv_w^T   [8192 x 3072 x 1024], bf16 out
    gemm_kernel<1, 0, 0><<<dim3(3 * C_ / 128, BN_ / 256), 256, 0, stream>>>(
        lnbuf, wqkvT, nullptr, qkvbuf, BN_, 3 * C_, C_);

    // 4) split into Q*scale / K^T / V
    split_qkv_kernel<<<(BN_ * 3 * C_) / 256, 256, 0, stream>>>(qkvbuf, qb, ktb, vb);

    // 5) attention (flash-style, online softmax)
    attention_kernel<<<(B_ * H_ * (N_ / 16)) / 8, 256, 0, stream>>>(qb, ktb, vb, attno);

    // 6) proj = attn @ proj_w^T + proj_b   [8192 x 1024 x 1024], fp32 out
    gemm_kernel<0, 0, 1><<<dim3(C_ / 128, BN_ / 256), 256, 0, stream>>>(
        attno, wprojT, proj_b, projf, BN_, C_, C_);

    // 7) x1 = x + gamma1 * proj
    residual_kernel<<<(BN_ * C_) / 256, 256, 0, stream>>>(x, projf, gamma1, x1, BN_ * C_);

    // 8) LN2
    layernorm_kernel<<<BN_, 256, 0, stream>>>(x1, ln2_w, ln2_b, lnbuf);

    // 9) h = gelu(ln2 @ fc1_w^T + fc1_b)   [8192 x 4096 x 1024], bf16 out
    gemm_kernel<1, 1, 1><<<dim3(HID_ / 128, BN_ / 256), 256, 0, stream>>>(
        lnbuf, wfc1T, fc1_b, hbuf, BN_, HID_, C_);

    // 10) fc2 = h @ fc2_w^T + fc2_b        [8192 x 1024 x 4096], fp32 out
    gemm_kernel<0, 0, 1><<<dim3(C_ / 128, BN_ / 256), 256, 0, stream>>>(
        hbuf, wfc2T, fc2_b, fc2f, BN_, C_, HID_);

    // 11) out = x1 + gamma2 * fc2
    residual_kernel<<<(BN_ * C_) / 256, 256, 0, stream>>>(x1, fc2f, gamma2, out, BN_ * C_);
}
